// MEGNETModel_7842610282555
// MI455X (gfx1250) — matrix operand order repaired and checked by reading, not compiled, not run on hardware
//
#include <hip/hip_runtime.h>

typedef __attribute__((ext_vector_type(16))) _Float16 v16h;
typedef __attribute__((ext_vector_type(8)))  float    v8f;
typedef __attribute__((ext_vector_type(4)))  int      i4;

union V16H { v16h v; i4 q[2]; };

// Packed f16 weight offsets (in halfs) inside the WPK region.
static constexpr int OFF_W1E1 = 0;      // [160][64]
static constexpr int OFF_W1E2 = 10240;  // [64][32]
static constexpr int OFF_W1V1 = 12288;  // [64][64]
static constexpr int OFF_W1V2 = 16384;  // [64][32]
static constexpr int OFF_W2E1 = 18432;  // [128][64]
static constexpr int OFF_W2E2 = 26624;  // [64][32]
static constexpr int OFF_W2V1 = 28672;  // [96][64]
static constexpr int OFF_W2V2 = 34816;  // [64][32]
static constexpr int WPK_HALFS = 36864;

// ---------- WMMA helpers (CDNA5 16x16x32 f16 layouts) ----------
// A (16-bit, 16xK row-major in LDS): lane m=l&15, K-half = l>>4.
__device__ __forceinline__ v16h lds_load_a(const _Float16* f, int ldk, int k0, int lane) {
  int m = lane & 15;
  int kh = (lane >> 4) << 3;
  const _Float16* p = f + m * ldk + k0 + kh;
  V16H a;
  a.q[0] = *(const i4*)p;
  a.q[1] = *(const i4*)(p + 16);
  return a.v;
}
// B (16-bit, 32x16 tile of row-major [Kp][Np] f16 in global): lane n=l&15, K-half = l>>4 (same k order as A).
__device__ __forceinline__ v16h g_load_b(const _Float16* w, int Np, int kt, int nt, int lane) {
  const _Float16* p = w + (kt * 32 + ((lane >> 4) << 3)) * Np + nt * 16 + (lane & 15);
  v16h b;
  for (int s = 0; s < 16; ++s) b[s] = p[((s & 7) + ((s >> 3) << 4)) * Np];
  return b;
}
__device__ __forceinline__ v8f wmma_f16(v16h a, v16h b, v8f c) {
  return __builtin_amdgcn_wmma_f32_16x16x32_f16(false, a, false, b, (short)0, c, false, false);
}

__device__ __forceinline__ unsigned fenc(float x) {
  unsigned u = __float_as_uint(x);
  return (u & 0x80000000u) ? ~u : (u | 0x80000000u);
}
__device__ __forceinline__ float fdec(unsigned u) {
  return (u & 0x80000000u) ? __uint_as_float(u ^ 0x80000000u) : __uint_as_float(~u);
}

// ---------- weight packing ----------
__global__ void pack_weights(const float* s0, const float* s1, const float* s2, const float* s3,
                             const float* s4, const float* s5, const float* s6, const float* s7,
                             _Float16* wpk) {
  const int Ks[8] = {134, 64, 50, 64, 128, 64, 96, 64};
  const int Kp[8] = {160, 64, 64, 64, 128, 64, 96, 64};
  const int Np[8] = {64, 32, 64, 32, 64, 32, 64, 32};
  const int Of[8] = {OFF_W1E1, OFF_W1E2, OFF_W1V1, OFF_W1V2, OFF_W2E1, OFF_W2E2, OFF_W2V1, OFF_W2V2};
  int j = blockIdx.x;
  const float* s = j == 0 ? s0 : j == 1 ? s1 : j == 2 ? s2 : j == 3 ? s3
                 : j == 4 ? s4 : j == 5 ? s5 : j == 6 ? s6 : s7;
  int total = Kp[j] * Np[j];
  for (int i = threadIdx.x; i < total; i += blockDim.x) {
    int k = i / Np[j], n = i % Np[j];
    wpk[Of[j] + i] = (_Float16)((k < Ks[j]) ? s[k * Np[j] + n] : 0.0f);
  }
}

// ---------- small prep kernels ----------
__global__ void embed_kernel(const int* __restrict__ x, const float* __restrict__ emb,
                             _Float16* __restrict__ v0h, int N) {
  int idx = blockIdx.x * blockDim.x + threadIdx.x;
  if (idx >= N * 16) return;
  int n = idx >> 4, c = idx & 15;
  v0h[idx] = (_Float16)emb[x[n] * 16 + c];
}

__global__ void ebatch_init(const int* __restrict__ src, const int* __restrict__ dst,
                            const int* __restrict__ batch, int* __restrict__ eb,
                            float* ncnt, float* gec, int E) {
  int i = blockIdx.x * blockDim.x + threadIdx.x;
  if (i >= E) return;
  int b = batch[src[i]];
  eb[i] = b;
  atomicAdd(&gec[b], 1.0f);
  atomicAdd(&ncnt[dst[i]], 1.0f);
}

__global__ void node_cnt_kernel(const int* __restrict__ batch, float* gvc, int N) {
  int n = blockIdx.x * blockDim.x + threadIdx.x;
  if (n >= N) return;
  atomicAdd(&gvc[batch[n]], 1.0f);
}

// ---------- edge MLP, block 1: [vsrc16|vdst16|rbf100|u2] -> 64 -> 32 ----------
__global__ __launch_bounds__(128) void edge_mlp1(
    const int* __restrict__ src, const int* __restrict__ dst,
    const float* __restrict__ eattr, const float* __restrict__ state,
    const int* __restrict__ ebatch, const _Float16* __restrict__ v0h,
    const _Float16* __restrict__ wpk,
    const float* __restrict__ be1, const float* __restrict__ be2,
    _Float16* __restrict__ e1h, int E) {
  __shared__ __align__(16) _Float16 sF[4][16][160];
  int wave = threadIdx.x >> 5, lane = threadIdx.x & 31;
  int tile = blockIdx.x * 4 + wave;
  int e0 = tile * 16;
  if (e0 >= E) return;
  int m = lane & 15, hsel = lane >> 4;
  int e = e0 + m; if (e >= E) e = E - 1;
  _Float16* frow = &sF[wave][m][0];
  float dist = eattr[e];
  if (hsel == 0) {
    int s = src[e], d = dst[e];
    const i4* vs = (const i4*)(v0h + (size_t)s * 16);
    const i4* vd = (const i4*)(v0h + (size_t)d * 16);
    i4* fq = (i4*)frow;
    fq[0] = vs[0]; fq[1] = vs[1]; fq[2] = vd[0]; fq[3] = vd[1];
    for (int c = 0; c < 48; ++c) {
      float ck = 5.0f * (float)c / 99.0f;
      float t = dist - ck;
      frow[32 + c] = (_Float16)__expf(-t * t * 4.0f);
    }
  } else {
    for (int c = 48; c < 100; ++c) {
      float ck = 5.0f * (float)c / 99.0f;
      float t = dist - ck;
      frow[32 + c] = (_Float16)__expf(-t * t * 4.0f);
    }
    int eb = ebatch[e];
    frow[132] = (_Float16)state[eb * 2 + 0];
    frow[133] = (_Float16)state[eb * 2 + 1];
    for (int k = 134; k < 160; ++k) frow[k] = (_Float16)0.0f;
  }
  asm volatile("s_wait_dscnt 0x0" ::: "memory");
  v8f acc[4] = {};
  for (int kt = 0; kt < 5; ++kt) {
    v16h a = lds_load_a(&sF[wave][0][0], 160, kt * 32, lane);
    for (int nt = 0; nt < 4; ++nt)
      acc[nt] = wmma_f16(a, g_load_b(wpk + OFF_W1E1, 64, kt, nt, lane), acc[nt]);
  }
  _Float16* hb = &sF[wave][0][0];
  int n0 = lane & 15, rb = hsel * 8;
  for (int nt = 0; nt < 4; ++nt) {
    float bb = be1[nt * 16 + n0];
    for (int r = 0; r < 8; ++r) {
      float h = acc[nt][r] + bb;
      hb[(rb + r) * 64 + nt * 16 + n0] = (_Float16)(h > 0.f ? h : 0.f);
    }
  }
  asm volatile("s_wait_dscnt 0x0" ::: "memory");
  v8f a2[2] = {};
  for (int kt = 0; kt < 2; ++kt) {
    v16h a = lds_load_a(hb, 64, kt * 32, lane);
    for (int nt = 0; nt < 2; ++nt)
      a2[nt] = wmma_f16(a, g_load_b(wpk + OFF_W1E2, 32, kt, nt, lane), a2[nt]);
  }
  for (int nt = 0; nt < 2; ++nt) {
    float bb = be2[nt * 16 + n0];
    for (int r = 0; r < 8; ++r) {
      float v = a2[nt][r] + bb;
      hb[(rb + r) * 32 + nt * 16 + n0] = (_Float16)(v > 0.f ? v : 0.f);
    }
  }
  asm volatile("s_wait_dscnt 0x0" ::: "memory");
  int m2 = lane >> 1, q = lane & 1;
  int er = e0 + m2;
  if (er < E) {
    const i4* ps = (const i4*)(hb + m2 * 32 + q * 16);
    i4 d0 = ps[0], d1 = ps[1];
    i4* pd = (i4*)(e1h + (size_t)er * 32 + q * 16);
    pd[0] = d0; pd[1] = d1;
  }
}

// ---------- edge MLP, block 2: [vsrc32|vdst32|e32|u32] -> 64 -> 32 ----------
__global__ __launch_bounds__(128) void edge_mlp2(
    const int* __restrict__ src, const int* __restrict__ dst,
    const int* __restrict__ ebatch, const _Float16* __restrict__ v1h,
    const _Float16* __restrict__ e1h, const _Float16* __restrict__ u1h,
    const _Float16* __restrict__ wpk,
    const float* __restrict__ be1, const float* __restrict__ be2,
    _Float16* __restrict__ e2h, int E) {
  __shared__ __align__(16) _Float16 sF[4][16][128];
  int wave = threadIdx.x >> 5, lane = threadIdx.x & 31;
  int tile = blockIdx.x * 4 + wave;
  int e0 = tile * 16;
  if (e0 >= E) return;
  int m = lane & 15, hsel = lane >> 4;
  int e = e0 + m; if (e >= E) e = E - 1;
  _Float16* frow = &sF[wave][m][0];
  i4* fq = (i4*)frow;
  if (hsel == 0) {
    int s = src[e], d = dst[e];
    const i4* vs = (const i4*)(v1h + (size_t)s * 32);
    const i4* vd = (const i4*)(v1h + (size_t)d * 32);
    fq[0] = vs[0]; fq[1] = vs[1]; fq[2] = vs[2]; fq[3] = vs[3];
    fq[4] = vd[0]; fq[5] = vd[1]; fq[6] = vd[2]; fq[7] = vd[3];
  } else {
    const i4* ee = (const i4*)(e1h + (size_t)e * 32);
    const i4* uu = (const i4*)(u1h + (size_t)ebatch[e] * 32);
    fq[8] = ee[0]; fq[9] = ee[1]; fq[10] = ee[2]; fq[11] = ee[3];
    fq[12] = uu[0]; fq[13] = uu[1]; fq[14] = uu[2]; fq[15] = uu[3];
  }
  asm volatile("s_wait_dscnt 0x0" ::: "memory");
  v8f acc[4] = {};
  for (int kt = 0; kt < 4; ++kt) {
    v16h a = lds_load_a(&sF[wave][0][0], 128, kt * 32, lane);
    for (int nt = 0; nt < 4; ++nt)
      acc[nt] = wmma_f16(a, g_load_b(wpk + OFF_W2E1, 64, kt, nt, lane), acc[nt]);
  }
  _Float16* hb = &sF[wave][0][0];
  int n0 = lane & 15, rb = hsel * 8;
  for (int nt = 0; nt < 4; ++nt) {
    float bb = be1[nt * 16 + n0];
    for (int r = 0; r < 8; ++r) {
      float h = acc[nt][r] + bb;
      hb[(rb + r) * 64 + nt * 16 + n0] = (_Float16)(h > 0.f ? h : 0.f);
    }
  }
  asm volatile("s_wait_dscnt 0x0" ::: "memory");
  v8f a2[2] = {};
  for (int kt = 0; kt < 2; ++kt) {
    v16h a = lds_load_a(hb, 64, kt * 32, lane);
    for (int nt = 0; nt < 2; ++nt)
      a2[nt] = wmma_f16(a, g_load_b(wpk + OFF_W2E2, 32, kt, nt, lane), a2[nt]);
  }
  for (int nt = 0; nt < 2; ++nt) {
    float bb = be2[nt * 16 + n0];
    for (int r = 0; r < 8; ++r) {
      float v = a2[nt][r] + bb;
      hb[(rb + r) * 32 + nt * 16 + n0] = (_Float16)(v > 0.f ? v : 0.f);
    }
  }
  asm volatile("s_wait_dscnt 0x0" ::: "memory");
  int m2 = lane >> 1, q = lane & 1;
  int er = e0 + m2;
  if (er < E) {
    const i4* ps = (const i4*)(hb + m2 * 32 + q * 16);
    i4 d0 = ps[0], d1 = ps[1];
    i4* pd = (i4*)(e2h + (size_t)er * 32 + q * 16);
    pd[0] = d0; pd[1] = d1;
  }
}

// ---------- node MLP, block 1: [v16|ev32|u2] -> 64 -> 32 ----------
__global__ __launch_bounds__(32) void node_mlp1(
    const int* __restrict__ batch, const _Float16* __restrict__ v0h,
    const float* __restrict__ nsum, const float* __restrict__ ncnt,
    const float* __restrict__ state, const _Float16* __restrict__ wpk,
    const float* __restrict__ bv1, const float* __restrict__ bv2,
    _Float16* __restrict__ v1h, int N) {
  __shared__ __align__(16) _Float16 sF[16][64];
  int lane = threadIdx.x;
  int n0t = blockIdx.x * 16;
  int m = lane & 15, hsel = lane >> 4;
  int n = n0t + m; if (n >= N) n = N - 1;
  _Float16* frow = &sF[m][0];
  float inv = 1.0f / fmaxf(ncnt[n], 1.0f);
  if (hsel == 0) {
    const i4* vv = (const i4*)(v0h + (size_t)n * 16);
    ((i4*)frow)[0] = vv[0]; ((i4*)frow)[1] = vv[1];
    for (int c = 0; c < 16; ++c) frow[16 + c] = (_Float16)(nsum[(size_t)n * 32 + c] * inv);
  } else {
    for (int c = 0; c < 16; ++c) frow[32 + c] = (_Float16)(nsum[(size_t)n * 32 + 16 + c] * inv);
    int b = batch[n];
    frow[48] = (_Float16)state[b * 2 + 0];
    frow[49] = (_Float16)state[b * 2 + 1];
    for (int k = 50; k < 64; ++k) frow[k] = (_Float16)0.0f;
  }
  asm volatile("s_wait_dscnt 0x0" ::: "memory");
  v8f acc[4] = {};
  for (int kt = 0; kt < 2; ++kt) {
    v16h a = lds_load_a(&sF[0][0], 64, kt * 32, lane);
    for (int nt = 0; nt < 4; ++nt)
      acc[nt] = wmma_f16(a, g_load_b(wpk + OFF_W1V1, 64, kt, nt, lane), acc[nt]);
  }
  _Float16* hb = &sF[0][0];
  int n0 = lane & 15, rb = hsel * 8;
  for (int nt = 0; nt < 4; ++nt) {
    float bb = bv1[nt * 16 + n0];
    for (int r = 0; r < 8; ++r) {
      float h = acc[nt][r] + bb;
      hb[(rb + r) * 64 + nt * 16 + n0] = (_Float16)(h > 0.f ? h : 0.f);
    }
  }
  asm volatile("s_wait_dscnt 0x0" ::: "memory");
  v8f a2[2] = {};
  for (int kt = 0; kt < 2; ++kt) {
    v16h a = lds_load_a(hb, 64, kt * 32, lane);
    for (int nt = 0; nt < 2; ++nt)
      a2[nt] = wmma_f16(a, g_load_b(wpk + OFF_W1V2, 32, kt, nt, lane), a2[nt]);
  }
  for (int nt = 0; nt < 2; ++nt) {
    float bb = bv2[nt * 16 + n0];
    for (int r = 0; r < 8; ++r) {
      float v = a2[nt][r] + bb;
      hb[(rb + r) * 32 + nt * 16 + n0] = (_Float16)(v > 0.f ? v : 0.f);
    }
  }
  asm volatile("s_wait_dscnt 0x0" ::: "memory");
  int m2 = lane >> 1, q = lane & 1;
  int nr = n0t + m2;
  if (nr < N) {
    const i4* ps = (const i4*)(hb + m2 * 32 + q * 16);
    i4 d0 = ps[0], d1 = ps[1];
    i4* pd = (i4*)(v1h + (size_t)nr * 32 + q * 16);
    pd[0] = d0; pd[1] = d1;
  }
}

// ---------- node MLP, block 2: [v32|ev32|u32] -> 64 -> 32 ----------
__global__ __launch_bounds__(32) void node_mlp2(
    const int* __restrict__ batch, const _Float16* __restrict__ v1h,
    const float* __restrict__ nsum, const float* __restrict__ ncnt,
    const _Float16* __restrict__ u1h, const _Float16* __restrict__ wpk,
    const float* __restrict__ bv1, const float* __restrict__ bv2,
    _Float16* __restrict__ v2h, int N) {
  __shared__ __align__(16) _Float16 sF[16][96];
  int lane = threadIdx.x;
  int n0t = blockIdx.x * 16;
  int m = lane & 15, hsel = lane >> 4;
  int n = n0t + m; if (n >= N) n = N - 1;
  _Float16* frow = &sF[m][0];
  float inv = 1.0f / fmaxf(ncnt[n], 1.0f);
  if (hsel == 0) {
    const i4* vv = (const i4*)(v1h + (size_t)n * 32);
    i4* fq = (i4*)frow;
    fq[0] = vv[0]; fq[1] = vv[1]; fq[2] = vv[2]; fq[3] = vv[3];
    for (int c = 0; c < 16; ++c) frow[32 + c] = (_Float16)(nsum[(size_t)n * 32 + c] * inv);
  } else {
    for (int c = 0; c < 16; ++c) frow[48 + c] = (_Float16)(nsum[(size_t)n * 32 + 16 + c] * inv);
    int b = batch[n];
    const i4* uu = (const i4*)(u1h + (size_t)b * 32);
    i4* fq = (i4*)frow;
    fq[4] = uu[0]; fq[5] = uu[1]; fq[6] = uu[2]; fq[7] = uu[3];
  }
  asm volatile("s_wait_dscnt 0x0" ::: "memory");
  v8f acc[4] = {};
  for (int kt = 0; kt < 3; ++kt) {
    v16h a = lds_load_a(&sF[0][0], 96, kt * 32, lane);
    for (int nt = 0; nt < 4; ++nt)
      acc[nt] = wmma_f16(a, g_load_b(wpk + OFF_W2V1, 64, kt, nt, lane), acc[nt]);
  }
  _Float16* hb = &sF[0][0];
  int n0 = lane & 15, rb = hsel * 8;
  for (int nt = 0; nt < 4; ++nt) {
    float bb = bv1[nt * 16 + n0];
    for (int r = 0; r < 8; ++r) {
      float h = acc[nt][r] + bb;
      hb[(rb + r) * 64 + nt * 16 + n0] = (_Float16)(h > 0.f ? h : 0.f);
    }
  }
  asm volatile("s_wait_dscnt 0x0" ::: "memory");
  v8f a2[2] = {};
  for (int kt = 0; kt < 2; ++kt) {
    v16h a = lds_load_a(hb, 64, kt * 32, lane);
    for (int nt = 0; nt < 2; ++nt)
      a2[nt] = wmma_f16(a, g_load_b(wpk + OFF_W2V2, 32, kt, nt, lane), a2[nt]);
  }
  for (int nt = 0; nt < 2; ++nt) {
    float bb = bv2[nt * 16 + n0];
    for (int r = 0; r < 8; ++r) {
      float v = a2[nt][r] + bb;
      hb[(rb + r) * 32 + nt * 16 + n0] = (_Float16)(v > 0.f ? v : 0.f);
    }
  }
  asm volatile("s_wait_dscnt 0x0" ::: "memory");
  int m2 = lane >> 1, q = lane & 1;
  int nr = n0t + m2;
  if (nr < N) {
    const i4* ps = (const i4*)(hb + m2 * 32 + q * 16);
    i4 d0 = ps[0], d1 = ps[1];
    i4* pd = (i4*)(v2h + (size_t)nr * 32 + q * 16);
    pd[0] = d0; pd[1] = d1;
  }
}

// ---------- segment accumulators ----------
__global__ void edge_accum(const _Float16* __restrict__ eh, const int* __restrict__ dst,
                           const int* __restrict__ eb, float* nsum, float* ges, int E) {
  int i = blockIdx.x * blockDim.x + threadIdx.x;
  if (i >= E) return;
  int d = dst[i], b = eb[i];
  for (int c = 0; c < 32; ++c) {
    float v = (float)eh[(size_t)i * 32 + c];
    atomicAdd(&nsum[(size_t)d * 32 + c], v);
    atomicAdd(&ges[b * 32 + c], v);
  }
}
__global__ void node_accum(const _Float16* __restrict__ vh, const int* __restrict__ batch,
                           float* gvs, int N) {
  int i = blockIdx.x * blockDim.x + threadIdx.x;
  if (i >= N) return;
  int b = batch[i];
  for (int c = 0; c < 32; ++c)
    atomicAdd(&gvs[b * 32 + c], (float)vh[(size_t)i * 32 + c]);
}

// ---------- state update MLP (scalar, B=512) ----------
__global__ __launch_bounds__(96) void state_mlp(
    const float* __restrict__ gES, const float* __restrict__ gEC,
    const float* __restrict__ gVS, const float* __restrict__ gVC,
    const float* __restrict__ uin, int du,
    const float* __restrict__ Wu1, const float* __restrict__ bu1,
    const float* __restrict__ Wu2, const float* __restrict__ bu2,
    float* uout, _Float16* uouth) {
  __shared__ float z[96], hh[64];
  int b = blockIdx.x, t = threadIdx.x;
  int K = 64 + du;
  if (t < 32) z[t] = gES[b * 32 + t] / fmaxf(gEC[b], 1.0f);
  else if (t < 64) z[t] = gVS[b * 32 + (t - 32)] / fmaxf(gVC[b], 1.0f);
  else z[t] = (t - 64 < du) ? uin[b * du + (t - 64)] : 0.0f;
  __syncthreads();
  if (t < 64) {
    float a = bu1[t];
    for (int k = 0; k < K; ++k) a += z[k] * Wu1[k * 64 + t];
    hh[t] = fmaxf(a, 0.0f);
  }
  __syncthreads();
  if (t < 32) {
    float a = bu2[t];
    for (int k = 0; k < 64; ++k) a += hh[k] * Wu2[k * 32 + t];
    a = fmaxf(a, 0.0f);
    uout[b * 32 + t] = a;
    if (uouth) uouth[b * 32 + t] = (_Float16)a;
  }
}

// ---------- set2set ----------
__global__ __launch_bounds__(128) void s2s_lstm(
    const float* __restrict__ Wih, const float* __restrict__ Whh,
    const float* __restrict__ bih, const float* __restrict__ bhh,
    float* h, float* c, const float* __restrict__ qstar,
    float* r, unsigned* mx, float* sm) {
  __shared__ float qs[64], hs[32], gs[128];
  int b = blockIdx.x, t = threadIdx.x;
  if (t < 64) qs[t] = qstar[b * 64 + t];
  if (t < 32) hs[t] = h[b * 32 + t];
  __syncthreads();
  float g = bih[t] + bhh[t];
  for (int k = 0; k < 64; ++k) g += qs[k] * Wih[k * 128 + t];
  for (int k = 0; k < 32; ++k) g += hs[k] * Whh[k * 128 + t];
  gs[t] = g;
  __syncthreads();
  if (t < 32) {
    float gi = gs[t], gf = gs[32 + t], gg = gs[64 + t], go = gs[96 + t];
    float si = 1.0f / (1.0f + __expf(-gi));
    float sf = 1.0f / (1.0f + __expf(-gf));
    float so = 1.0f / (1.0f + __expf(-go));
    float cc = sf * c[b * 32 + t] + si * tanhf(gg);
    c[b * 32 + t] = cc;
    h[b * 32 + t] = so * tanhf(cc);
    r[b * 32 + t] = 0.0f;
  }
  if (t == 0) { mx[b] = 0u; sm[b] = 0.0f; }
}

__global__ void s2s_energy(const _Float16* __restrict__ xf, const int* __restrict__ ids,
                           const float* __restrict__ h, float* p, unsigned* mx, int n) {
  int i = blockIdx.x * blockDim.x + threadIdx.x;
  if (i >= n) return;
  int b = ids[i];
  float en = 0.0f;
  for (int c = 0; c < 32; ++c) en += (float)xf[(size_t)i * 32 + c] * h[b * 32 + c];
  p[i] = en;
  atomicMax(&mx[b], fenc(en));
}
__global__ void s2s_expsum(const int* __restrict__ ids, const unsigned* __restrict__ mx,
                           float* p, float* s, int n) {
  int i = blockIdx.x * blockDim.x + threadIdx.x;
  if (i >= n) return;
  float mm = fdec(mx[ids[i]]);
  float pv = __expf(p[i] - mm);
  p[i] = pv;
  atomicAdd(&s[ids[i]], pv);
}
__global__ void s2s_raccum(const _Float16* __restrict__ xf, const int* __restrict__ ids,
                           const float* __restrict__ p, const float* __restrict__ s,
                           float* r, int n) {
  int i = blockIdx.x * blockDim.x + threadIdx.x;
  if (i >= n) return;
  int b = ids[i];
  float a = p[i] / (s[b] + 1e-12f);
  for (int c = 0; c < 32; ++c)
    atomicAdd(&r[b * 32 + c], a * (float)xf[(size_t)i * 32 + c]);
}
__global__ void s2s_qupdate(const float* __restrict__ h, const float* __restrict__ r,
                            float* qstar, int B) {
  int idx = blockIdx.x * blockDim.x + threadIdx.x;
  if (idx >= B * 32) return;
  int b = idx >> 5, k = idx & 31;
  qstar[b * 64 + k] = h[idx];
  qstar[b * 64 + 32 + k] = r[idx];
}

// ---------- head with batch-norm ----------
__global__ void head1(const float* __restrict__ qn, const float* __restrict__ qe,
                      const float* __restrict__ u2, const float* __restrict__ W1,
                      const float* __restrict__ b1, float* z1, float* cs, int B) {
  int idx = blockIdx.x * blockDim.x + threadIdx.x;
  if (idx >= B * 32) return;
  int b = idx >> 5, j = idx & 31;
  float acc = b1[j];
  for (int k = 0; k < 64; ++k) acc += qn[b * 64 + k] * W1[k * 32 + j];
  for (int k = 0; k < 64; ++k) acc += qe[b * 64 + k] * W1[(64 + k) * 32 + j];
  for (int k = 0; k < 32; ++k) acc += u2[b * 32 + k] * W1[(128 + k) * 32 + j];
  z1[idx] = acc;
  atomicAdd(&cs[j], acc);
  atomicAdd(&cs[32 + j], acc * acc);
}
__global__ void head2(const float* __restrict__ z1, const float* __restrict__ cs,
                      const float* __restrict__ g1, const float* __restrict__ bt1,
                      const float* __restrict__ W2, const float* __restrict__ b2,
                      float* z2, float* cs2, int B, float fB) {
  int idx = blockIdx.x * blockDim.x + threadIdx.x;
  if (idx >= B * 16) return;
  int b = idx >> 4, j = idx & 15;
  float acc = b2[j];
  for (int k = 0; k < 32; ++k) {
    float mu = cs[k] / fB;
    float var = cs[32 + k] / fB - mu * mu;
    float zn = (z1[b * 32 + k] - mu) * rsqrtf(var + 1e-5f) * g1[k] + bt1[k];
    zn = fmaxf(zn, 0.0f);
    acc += zn * W2[k * 16 + j];
  }
  z2[idx] = acc;
  atomicAdd(&cs2[j], acc);
  atomicAdd(&cs2[16 + j], acc * acc);
}
__global__ void head3(const float* __restrict__ z2, const float* __restrict__ cs2,
                      const float* __restrict__ g2, const float* __restrict__ bt2,
                      const float* __restrict__ Wo, const float* __restrict__ bo,
                      float* out, int B, float fB) {
  int b = blockIdx.x * blockDim.x + threadIdx.x;
  if (b >= B) return;
  float acc = bo[0];
  for (int k = 0; k < 16; ++k) {
    float mu = cs2[k] / fB;
    float var = cs2[16 + k] / fB - mu * mu;
    float zn = (z2[b * 16 + k] - mu) * rsqrtf(var + 1e-5f) * g2[k] + bt2[k];
    zn = fmaxf(zn, 0.0f);
    acc += zn * Wo[k];
  }
  out[b] = acc;
}

// =======================================================================
extern "C" void kernel_launch(void* const* d_in, const int* in_sizes, int n_in,
                              void* d_out, int out_size, void* d_ws, size_t ws_size,
                              hipStream_t stream) {
  (void)n_in; (void)out_size; (void)ws_size;
  const int N = in_sizes[0];
  const int E = in_sizes[2];
  const int B = in_sizes[3] / 2;

  const int*   x     = (const int*)d_in[0];
  const int*   eidx  = (const int*)d_in[1];
  const float* eattr = (const float*)d_in[2];
  const float* state = (const float*)d_in[3];
  const int*   batch = (const int*)d_in[4];
  const float* emb   = (const float*)d_in[5];
  const float* b1We1 = (const float*)d_in[6];  const float* b1be1 = (const float*)d_in[7];
  const float* b1We2 = (const float*)d_in[8];  const float* b1be2 = (const float*)d_in[9];
  const float* b1Wv1 = (const float*)d_in[10]; const float* b1bv1 = (const float*)d_in[11];
  const float* b1Wv2 = (const float*)d_in[12]; const float* b1bv2 = (const float*)d_in[13];
  const float* b1Wu1 = (const float*)d_in[14]; const float* b1bu1 = (const float*)d_in[15];
  const float* b1Wu2 = (const float*)d_in[16]; const float* b1bu2 = (const float*)d_in[17];
  const float* b2We1 = (const float*)d_in[18]; const float* b2be1 = (const float*)d_in[19];
  const float* b2We2 = (const float*)d_in[20]; const float* b2be2 = (const float*)d_in[21];
  const float* b2Wv1 = (const float*)d_in[22]; const float* b2bv1 = (const float*)d_in[23];
  const float* b2Wv2 = (const float*)d_in[24]; const float* b2bv2 = (const float*)d_in[25];
  const float* b2Wu1 = (const float*)d_in[26]; const float* b2bu1 = (const float*)d_in[27];
  const float* b2Wu2 = (const float*)d_in[28]; const float* b2bu2 = (const float*)d_in[29];
  const float* nWih = (const float*)d_in[30]; const float* nWhh = (const float*)d_in[31];
  const float* nbih = (const float*)d_in[32]; const float* nbhh = (const float*)d_in[33];
  const float* eWih = (const float*)d_in[34]; const float* eWhh = (const float*)d_in[35];
  const float* ebih = (const float*)d_in[36]; const float* ebhh = (const float*)d_in[37];
  const float* hW1 = (const float*)d_in[38]; const float* hb1 = (const float*)d_in[39];
  const float* hg1 = (const float*)d_in[40]; const float* hbt1 = (const float*)d_in[41];
  const float* hW2 = (const float*)d_in[42]; const float* hb2 = (const float*)d_in[43];
  const float* hg2 = (const float*)d_in[44]; const float* hbt2 = (const float*)d_in[45];
  const float* hWo = (const float*)d_in[46]; const float* hbo = (const float*)d_in[47];

  const int* src = eidx;
  const int* dst = eidx + E;
  float* out = (float*)d_out;

  char* ws = (char*)d_ws;
  size_t off = 0;
  auto alloc = [&](size_t bytes) { size_t r = off; off += (bytes + 255) & ~(size_t)255; return r; };
  size_t oWPK = alloc((size_t)WPK_HALFS * 2);
  size_t oV0  = alloc((size_t)N * 16 * 2);
  size_t oV1  = alloc((size_t)N * 32 * 2);
  size_t oV2  = alloc((size_t)N * 32 * 2);
  size_t oE1  = alloc((size_t)E * 32 * 2);
  size_t oE2  = alloc((size_t)E * 32 * 2);
  size_t oEB  = alloc((size_t)E * 4);
  size_t oCNT = off;
  size_t oNC  = alloc((size_t)N * 4);
  size_t oGEC = alloc((size_t)B * 4);
  size_t oGVC = alloc((size_t)B * 4);
  size_t zCNT = off - oCNT;
  size_t oSUM = off;
  size_t oNS  = alloc((size_t)N * 32 * 4);
  size_t oGES = alloc((size_t)B * 32 * 4);
  size_t oGVS = alloc((size_t)B * 32 * 4);
  size_t zSUM = off - oSUM;
  size_t oU1  = alloc((size_t)B * 32 * 4);
  size_t oU1H = alloc((size_t)B * 32 * 2);
  size_t oU2  = alloc((size_t)B * 32 * 4);
  size_t oS2  = off;
  size_t oHN = alloc((size_t)B * 32 * 4), oCN = alloc((size_t)B * 32 * 4), oQN = alloc((size_t)B * 64 * 4);
  size_t oHE = alloc((size_t)B * 32 * 4), oCE = alloc((size_t)B * 32 * 4), oQE = alloc((size_t)B * 64 * 4);
  size_t zS2 = off - oS2;
  size_t oRN = alloc((size_t)B * 32 * 4), oMN = alloc((size_t)B * 4), oSN = alloc((size_t)B * 4), oPN = alloc((size_t)N * 4);
  size_t oRE = alloc((size_t)B * 32 * 4), oME = alloc((size_t)B * 4), oSE = alloc((size_t)B * 4), oPE = alloc((size_t)E * 4);
  size_t oZ1 = alloc((size_t)B * 32 * 4), oZ2 = alloc((size_t)B * 16 * 4);
  size_t oHS = off;
  size_t oCS1 = alloc(64 * 4);
  size_t oCS2 = alloc(32 * 4);
  size_t zHS = off - oHS;

  _Float16* WPK = (_Float16*)(ws + oWPK);
  _Float16* V0H = (_Float16*)(ws + oV0);
  _Float16* V1H = (_Float16*)(ws + oV1);
  _Float16* V2H = (_Float16*)(ws + oV2);
  _Float16* E1H = (_Float16*)(ws + oE1);
  _Float16* E2H = (_Float16*)(ws + oE2);
  int* EB = (int*)(ws + oEB);
  float* NC = (float*)(ws + oNC);
  float* GEC = (float*)(ws + oGEC);
  float* GVC = (float*)(ws + oGVC);
  float* NS = (float*)(ws + oNS);
  float* GES = (float*)(ws + oGES);
  float* GVS = (float*)(ws + oGVS);
  float* U1 = (float*)(ws + oU1);
  _Float16* U1H = (_Float16*)(ws + oU1H);
  float* U2 = (float*)(ws + oU2);
  float* HN = (float*)(ws + oHN); float* CN = (float*)(ws + oCN); float* QN = (float*)(ws + oQN);
  float* HE = (float*)(ws + oHE); float* CE = (float*)(ws + oCE); float* QE = (float*)(ws + oQE);
  float* RN = (float*)(ws + oRN); unsigned* MN = (unsigned*)(ws + oMN); float* SN = (float*)(ws + oSN);
  float* PN = (float*)(ws + oPN);
  float* RE = (float*)(ws + oRE); unsigned* ME = (unsigned*)(ws + oME); float* SE = (float*)(ws + oSE);
  float* PE = (float*)(ws + oPE);
  float* Z1 = (float*)(ws + oZ1); float* Z2 = (float*)(ws + oZ2);
  float* CS1 = (float*)(ws + oCS1); float* CS2 = (float*)(ws + oCS2);

  // zero accumulators (graph-capture safe)
  hipMemsetAsync(ws + oCNT, 0, zCNT, stream);
  hipMemsetAsync(ws + oSUM, 0, zSUM, stream);
  hipMemsetAsync(ws + oS2, 0, zS2, stream);
  hipMemsetAsync(ws + oHS, 0, zHS, stream);

  const int gE = (E + 255) / 256;
  const int gN = (N + 255) / 256;
  const int eTiles = (E + 15) / 16;
  const int nTiles = (N + 15) / 16;

  pack_weights<<<8, 256, 0, stream>>>(b1We1, b1We2, b1Wv1, b1Wv2, b2We1, b2We2, b2Wv1, b2Wv2, WPK);
  embed_kernel<<<(N * 16 + 255) / 256, 256, 0, stream>>>(x, emb, V0H, N);
  ebatch_init<<<gE, 256, 0, stream>>>(src, dst, batch, EB, NC, GEC, E);
  node_cnt_kernel<<<gN, 256, 0, stream>>>(batch, GVC, N);

  // ---- block 1 ----
  edge_mlp1<<<(eTiles + 3) / 4, 128, 0, stream>>>(src, dst, eattr, state, EB, V0H, WPK,
                                                  b1be1, b1be2, E1H, E);
  edge_accum<<<gE, 256, 0, stream>>>(E1H, dst, EB, NS, GES, E);
  node_mlp1<<<nTiles, 32, 0, stream>>>(batch, V0H, NS, NC, state, WPK, b1bv1, b1bv2, V1H, N);
  node_accum<<<gN, 256, 0, stream>>>(V1H, batch, GVS, N);
  state_mlp<<<B, 96, 0, stream>>>(GES, GEC, GVS, GVC, state, 2, b1Wu1, b1bu1, b1Wu2, b1bu2, U1, U1H);

  // ---- block 2 ----
  hipMemsetAsync(ws + oSUM, 0, zSUM, stream);
  edge_mlp2<<<(eTiles + 3) / 4, 128, 0, stream>>>(src, dst, EB, V1H, E1H, U1H, WPK,
                                                  b2be1, b2be2, E2H, E);
  edge_accum<<<gE, 256, 0, stream>>>(E2H, dst, EB, NS, GES, E);
  node_mlp2<<<nTiles, 32, 0, stream>>>(batch, V1H, NS, NC, U1H, WPK, b2bv1, b2bv2, V2H, N);
  node_accum<<<gN, 256, 0, stream>>>(V2H, batch, GVS, N);
  state_mlp<<<B, 96, 0, stream>>>(GES, GEC, GVS, GVC, U1, 32, b2Wu1, b2bu1, b2Wu2, b2bu2, U2, (_Float16*)nullptr);

  // ---- set2set over nodes ----
  for (int step = 0; step < 3; ++step) {
    s2s_lstm<<<B, 128, 0, stream>>>(nWih, nWhh, nbih, nbhh, HN, CN, QN, RN, MN, SN);
    s2s_energy<<<gN, 256, 0, stream>>>(V2H, batch, HN, PN, MN, N);
    s2s_expsum<<<gN, 256, 0, stream>>>(batch, MN, PN, SN, N);
    s2s_raccum<<<gN, 256, 0, stream>>>(V2H, batch, PN, SN, RN, N);
    s2s_qupdate<<<(B * 32 + 255) / 256, 256, 0, stream>>>(HN, RN, QN, B);
  }
  // ---- set2set over edges ----
  for (int step = 0; step < 3; ++step) {
    s2s_lstm<<<B, 128, 0, stream>>>(eWih, eWhh, ebih, ebhh, HE, CE, QE, RE, ME, SE);
    s2s_energy<<<gE, 256, 0, stream>>>(E2H, EB, HE, PE, ME, E);
    s2s_expsum<<<gE, 256, 0, stream>>>(EB, ME, PE, SE, E);
    s2s_raccum<<<gE, 256, 0, stream>>>(E2H, EB, PE, SE, RE, E);
    s2s_qupdate<<<(B * 32 + 255) / 256, 256, 0, stream>>>(HE, RE, QE, B);
  }

  // ---- head ----
  head1<<<(B * 32 + 255) / 256, 256, 0, stream>>>(QN, QE, U2, hW1, hb1, Z1, CS1, B);
  head2<<<(B * 16 + 255) / 256, 256, 0, stream>>>(Z1, CS1, hg1, hbt1, hW2, hb2, Z2, CS2, B, (float)B);
  head3<<<(B + 255) / 256, 256, 0, stream>>>(Z2, CS2, hg2, hbt2, hWo, hbo, out, B, (float)B);
}
